// TacGen_51470888075362
// MI455X (gfx1250) — matrix operand order repaired and checked by reading, not compiled, not run on hardware
//
#include <hip/hip_runtime.h>
#include <hip/hip_fp16.h>

typedef __attribute__((ext_vector_type(16))) _Float16 v16h;
typedef __attribute__((ext_vector_type(8)))  _Float16 v8h;
typedef __attribute__((ext_vector_type(8)))  float    v8f;

#define WMMA16(A_, B_, C_) \
  __builtin_amdgcn_wmma_f32_16x16x32_f16(false, (A_), false, (B_), (short)0, (C_), false, false)

#define NEGF (-32767.0f)

// ---------------------------------------------------------------------------
// Fragment loaders (CDNA5 wave32 WMMA layouts, CDNA5 ISA §7.12.2)
// A 16x32 f16: lane L(<16) = row L, holds K=[8h..8h+7] and [16+8h..16+8h+7]
//              where h = lane>>4.  Two 16B loads.
// B 32x16 f16: lane = column, holds K=[8h..8h+7] and [16+8h..16+8h+7] (two 16B loads)
//              -> store B as Bt[N,K] row-major (i.e. transposed weights).
// C/D 16x16 f32: VGPR r, lane: row = r + 8*(lane>>4), col = lane&15.
// ---------------------------------------------------------------------------
static __device__ __forceinline__ v16h ld_a(const _Float16* p0, size_t lda,
                                            int row, int k0, int lane) {
  const _Float16* p = p0 + (size_t)(row + (lane & 15)) * lda
                         + (size_t)k0 + (size_t)((lane >> 4) << 3);
  v8h lo = *(const v8h*)p;
  v8h hi = *(const v8h*)(p + 16);
  v16h r;
#pragma unroll
  for (int i = 0; i < 8; ++i) { r[i] = lo[i]; r[i + 8] = hi[i]; }
  return r;
}

static __device__ __forceinline__ v16h ld_a_g(const _Float16* p0, size_t lda,
                                              int row, int k0, int lane, int rvalid) {
  int m = row + (lane & 15);
  v16h r;
  if (m < rvalid) {
    const _Float16* p = p0 + (size_t)m * lda + (size_t)k0 + (size_t)((lane >> 4) << 3);
    v8h lo = *(const v8h*)p;
    v8h hi = *(const v8h*)(p + 16);
#pragma unroll
    for (int i = 0; i < 8; ++i) { r[i] = lo[i]; r[i + 8] = hi[i]; }
  } else {
#pragma unroll
    for (int i = 0; i < 16; ++i) r[i] = (_Float16)0.0f;
  }
  return r;
}

static __device__ __forceinline__ v16h ld_b(const _Float16* p0, size_t ldb,
                                            int ncol, int k0, int lane) {
  const _Float16* p = p0 + (size_t)(ncol + (lane & 15)) * ldb
                         + (size_t)k0 + (size_t)((lane >> 4) << 3);
  v8h lo = *(const v8h*)p;
  v8h hi = *(const v8h*)(p + 16);
  v16h r;
#pragma unroll
  for (int i = 0; i < 8; ++i) { r[i] = lo[i]; r[i + 8] = hi[i]; }
  return r;
}

static __device__ __forceinline__ v16h ld_b_g(const _Float16* p0, size_t ldb,
                                              int ncol, int k0, int lane, int nvalid) {
  int n = ncol + (lane & 15);
  v16h r;
  if (n < nvalid) {
    const _Float16* p = p0 + (size_t)n * ldb + (size_t)k0 + (size_t)((lane >> 4) << 3);
    v8h lo = *(const v8h*)p;
    v8h hi = *(const v8h*)(p + 16);
#pragma unroll
    for (int i = 0; i < 8; ++i) { r[i] = lo[i]; r[i + 8] = hi[i]; }
  } else {
#pragma unroll
    for (int i = 0; i < 16; ++i) r[i] = (_Float16)0.0f;
  }
  return r;
}

// ---------------------------------------------------------------------------
// Generic GEMM core: wave computes a 32(M) x 64(N) tile, K in steps of 32.
// Block = 256 threads = 8 waves arranged 2(M) x 4(N): block tile 64 x 256.
// N is always 512 here -> gridDim.x = (M/64)*2.
// ---------------------------------------------------------------------------
static __device__ __forceinline__ void gemm_core(const _Float16* __restrict__ A, int lda,
                                                 const _Float16* __restrict__ Bt, int ldb,
                                                 int m0, int n0, int Ka, int lane,
                                                 v8f acc[2][4]) {
  v8f z = {};
#pragma unroll
  for (int i = 0; i < 2; ++i)
#pragma unroll
    for (int j = 0; j < 4; ++j) acc[i][j] = z;

  for (int k0 = 0; k0 < Ka; k0 += 32) {
    if (k0 + 32 < Ka) {
      __builtin_prefetch(A + (size_t)m0 * lda + k0 + 32, 0, 1);
      __builtin_prefetch(Bt + (size_t)n0 * ldb + k0 + 32, 0, 1);
    }
    v16h a0 = ld_a(A, (size_t)lda, m0,      k0, lane);
    v16h a1 = ld_a(A, (size_t)lda, m0 + 16, k0, lane);
    v16h b0 = ld_b(Bt, (size_t)ldb, n0,      k0, lane);
    v16h b1 = ld_b(Bt, (size_t)ldb, n0 + 16, k0, lane);
    v16h b2 = ld_b(Bt, (size_t)ldb, n0 + 32, k0, lane);
    v16h b3 = ld_b(Bt, (size_t)ldb, n0 + 48, k0, lane);
    acc[0][0] = WMMA16(a0, b0, acc[0][0]);
    acc[0][1] = WMMA16(a0, b1, acc[0][1]);
    acc[0][2] = WMMA16(a0, b2, acc[0][2]);
    acc[0][3] = WMMA16(a0, b3, acc[0][3]);
    acc[1][0] = WMMA16(a1, b0, acc[1][0]);
    acc[1][1] = WMMA16(a1, b1, acc[1][1]);
    acc[1][2] = WMMA16(a1, b2, acc[1][2]);
    acc[1][3] = WMMA16(a1, b3, acc[1][3]);
  }
}

#define GEMM_PROLOG()                               \
  int lane = threadIdx.x & 31;                      \
  int wv = threadIdx.x >> 5;                        \
  int mb = blockIdx.x >> 1;                         \
  int nb = blockIdx.x & 1;                          \
  int m0 = mb * 64 + (wv & 1) * 32;                 \
  int n0 = nb * 256 + (wv >> 1) * 64;               \
  v8f acc[2][4];                                    \
  gemm_core(A, lda, Bt, ldb, m0, n0, Ka, lane, acc);\
  int half = lane >> 4;                             \
  int c0 = lane & 15;

// out_f16 = relu(A @ Bt^T + bias)
__global__ __launch_bounds__(256) void k_gemm_relu_f16(
    const _Float16* __restrict__ A, int lda,
    const _Float16* __restrict__ Bt, int ldb,
    const float* __restrict__ bias,
    _Float16* __restrict__ C, int Ka) {
  GEMM_PROLOG();
#pragma unroll
  for (int i = 0; i < 2; ++i)
#pragma unroll
    for (int j = 0; j < 4; ++j)
#pragma unroll
      for (int r = 0; r < 8; ++r) {
        int row = m0 + i * 16 + r + half * 8;
        int col = n0 + j * 16 + c0;
        float v = acc[i][j][r] + bias[col];
        v = v > 0.0f ? v : 0.0f;
        C[(size_t)row * 512 + col] = (_Float16)v;
      }
}

// out_f32 = A @ Bt^T          (used for G1 = HS @ g_Wxs)
__global__ __launch_bounds__(256) void k_gemm_f32(
    const _Float16* __restrict__ A, int lda,
    const _Float16* __restrict__ Bt, int ldb,
    float* __restrict__ C, int Ka) {
  GEMM_PROLOG();
#pragma unroll
  for (int i = 0; i < 2; ++i)
#pragma unroll
    for (int j = 0; j < 4; ++j)
#pragma unroll
      for (int r = 0; r < 8; ++r) {
        int row = m0 + i * 16 + r + half * 8;
        int col = n0 + j * 16 + c0;
        C[(size_t)row * 512 + col] = acc[i][j][r];
      }
}

// z = sigmoid(G1 + HT @ g_Wxt + bxt);  Hf = z*HS + (1-z)*HT   (A = HT)
__global__ __launch_bounds__(256) void k_fuse(
    const _Float16* __restrict__ A, int lda,
    const _Float16* __restrict__ Bt, int ldb,
    const float* __restrict__ bias,
    const float* __restrict__ G1,
    const _Float16* __restrict__ HS,
    const _Float16* __restrict__ HT,
    _Float16* __restrict__ Hf, int Ka) {
  GEMM_PROLOG();
#pragma unroll
  for (int i = 0; i < 2; ++i)
#pragma unroll
    for (int j = 0; j < 4; ++j)
#pragma unroll
      for (int r = 0; r < 8; ++r) {
        int row = m0 + i * 16 + r + half * 8;
        int col = n0 + j * 16 + c0;
        size_t idx = (size_t)row * 512 + col;
        float a = acc[i][j][r] + G1[idx] + bias[col];
        float zg = 1.0f / (1.0f + __expf(-a));
        float hs = (float)HS[idx];
        float ht = (float)HT[idx];
        Hf[idx] = (_Float16)(zg * hs + (1.0f - zg) * ht);
      }
}

// out_f32 = X + A @ Bt^T + bias     (final residual)
__global__ __launch_bounds__(256) void k_final(
    const _Float16* __restrict__ A, int lda,
    const _Float16* __restrict__ Bt, int ldb,
    const float* __restrict__ bias,
    const float* __restrict__ X,
    float* __restrict__ Out, int Ka) {
  GEMM_PROLOG();
#pragma unroll
  for (int i = 0; i < 2; ++i)
#pragma unroll
    for (int j = 0; j < 4; ++j)
#pragma unroll
      for (int r = 0; r < 8; ++r) {
        int row = m0 + i * 16 + r + half * 8;
        int col = n0 + j * 16 + c0;
        size_t idx = (size_t)row * 512 + col;
        Out[idx] = X[idx] + acc[i][j][r] + bias[col];
      }
}

// ---------------------------------------------------------------------------
// Spatial attention: per (b*t, head, 128-row slab); one wave = 16 query rows.
// Flash-style online softmax over 512 keys, 32 keys per iteration.
// ---------------------------------------------------------------------------
__global__ __launch_bounds__(256) void k_sp_attn(
    const _Float16* __restrict__ Q, const _Float16* __restrict__ Kx,
    const _Float16* __restrict__ V, _Float16* __restrict__ O) {
  __shared__ alignas(32) _Float16 ldsP[8][16 * 32];
  __shared__ alignas(32) _Float16 ldsVt[8][64 * 32];
  int lane = threadIdx.x & 31;
  int wv = threadIdx.x >> 5;
  int bid = blockIdx.x;
  int rb = bid & 3;
  int hk = (bid >> 2) & 7;
  int bt = bid >> 5;                      // 0..47
  int m0 = rb * 128 + wv * 16;
  const _Float16* Qp = Q + (size_t)bt * 512 * 512 + hk * 64;
  const _Float16* Kp = Kx + (size_t)bt * 512 * 512 + hk * 64;
  const _Float16* Vp = V + (size_t)bt * 512 * 512 + hk * 64;

  v16h qa0 = ld_a(Qp, 512, m0, 0, lane);
  v16h qa1 = ld_a(Qp, 512, m0, 32, lane);

  v8f zero = {};
  v8f accv[4];
#pragma unroll
  for (int j = 0; j < 4; ++j) accv[j] = zero;
  float mrow[8], lrow[8];
#pragma unroll
  for (int r = 0; r < 8; ++r) { mrow[r] = -1e30f; lrow[r] = 0.0f; }
  const float scale = 0.125f;
  int half = lane >> 4;
  int c0 = lane & 15;

  for (int j0 = 0; j0 < 512; j0 += 32) {
    // scores: two 16x16 tiles, K=64 -> 2 WMMAs each
    v8f s0 = zero, s1 = zero;
    s0 = WMMA16(qa0, ld_b(Kp, 512, j0, 0, lane), s0);
    s0 = WMMA16(qa1, ld_b(Kp, 512, j0, 32, lane), s0);
    s1 = WMMA16(qa0, ld_b(Kp, 512, j0 + 16, 0, lane), s1);
    s1 = WMMA16(qa1, ld_b(Kp, 512, j0 + 16, 32, lane), s1);

    float mx[8];
#pragma unroll
    for (int r = 0; r < 8; ++r) {
      s0[r] *= scale; s1[r] *= scale;
      mx[r] = fmaxf(s0[r], s1[r]);
    }
#pragma unroll
    for (int d = 1; d < 16; d <<= 1)
#pragma unroll
      for (int r = 0; r < 8; ++r)
        mx[r] = fmaxf(mx[r], __shfl_xor(mx[r], d, 32));

    float p0[8], p1[8], rs[8];
#pragma unroll
    for (int r = 0; r < 8; ++r) {
      float nm = fmaxf(mrow[r], mx[r]);
      float corr = __expf(mrow[r] - nm);
      mrow[r] = nm;
      p0[r] = __expf(s0[r] - nm);
      p1[r] = __expf(s1[r] - nm);
      rs[r] = p0[r] + p1[r];
      lrow[r] *= corr;
#pragma unroll
      for (int j = 0; j < 4; ++j) accv[j][r] *= corr;
    }
#pragma unroll
    for (int d = 1; d < 16; d <<= 1)
#pragma unroll
      for (int r = 0; r < 8; ++r) rs[r] += __shfl_xor(rs[r], d, 32);
#pragma unroll
    for (int r = 0; r < 8; ++r) lrow[r] += rs[r];

    // stage P (C-layout -> A-layout via LDS)
#pragma unroll
    for (int r = 0; r < 8; ++r) {
      int row = r + half * 8;
      ldsP[wv][row * 32 + c0] = (_Float16)p0[r];
      ldsP[wv][row * 32 + 16 + c0] = (_Float16)p1[r];
    }
    // stage V^T: lane = key row, transpose into [dd][key]
    {
      const _Float16* vr = Vp + (size_t)(j0 + lane) * 512;
      v16h v0 = *(const v16h*)(vr);
      v16h v1 = *(const v16h*)(vr + 16);
      v16h v2 = *(const v16h*)(vr + 32);
      v16h v3 = *(const v16h*)(vr + 48);
#pragma unroll
      for (int e = 0; e < 16; ++e) {
        ldsVt[wv][(e) * 32 + lane] = v0[e];
        ldsVt[wv][(16 + e) * 32 + lane] = v1[e];
        ldsVt[wv][(32 + e) * 32 + lane] = v2[e];
        ldsVt[wv][(48 + e) * 32 + lane] = v3[e];
      }
    }
    asm volatile("s_wait_dscnt 0" ::: "memory");

    v16h pa = ld_a(&ldsP[wv][0], 32, 0, 0, lane);
#pragma unroll
    for (int df = 0; df < 4; ++df)
      accv[df] = WMMA16(pa, ld_b(&ldsVt[wv][0], 32, df * 16, 0, lane), accv[df]);
  }

  // epilogue: normalize by l, store f16
#pragma unroll
  for (int df = 0; df < 4; ++df)
#pragma unroll
    for (int r = 0; r < 8; ++r) {
      int row = m0 + r + half * 8;
      float v = accv[df][r] / lrow[r];
      O[((size_t)bt * 512 + row) * 512 + hk * 64 + df * 16 + c0] = (_Float16)v;
    }
}

// ---------------------------------------------------------------------------
// Temporal attention: one wave per (b, node, head); T=24 padded to 32, causal.
// ---------------------------------------------------------------------------
__global__ __launch_bounds__(256) void k_tm_attn(
    const _Float16* __restrict__ Q, const _Float16* __restrict__ Kx,
    const _Float16* __restrict__ V, _Float16* __restrict__ O) {
  __shared__ alignas(32) _Float16 ldsP[8][16 * 32];
  __shared__ alignas(32) _Float16 ldsVt[8][64 * 32];
  int lane = threadIdx.x & 31;
  int wv = threadIdx.x >> 5;
  int wg = blockIdx.x * 8 + wv;           // 0..8191
  int hk = wg & 7;
  int nd = (wg >> 3) & 511;
  int b = wg >> 12;
  const size_t tstride = (size_t)512 * 512;
  const _Float16* Qp = Q + ((size_t)b * 24 * 512 + nd) * 512 + hk * 64;
  const _Float16* Kp = Kx + ((size_t)b * 24 * 512 + nd) * 512 + hk * 64;
  const _Float16* Vp = V + ((size_t)b * 24 * 512 + nd) * 512 + hk * 64;
  _Float16* Op = O + ((size_t)b * 24 * 512 + nd) * 512 + hk * 64;

  int half = lane >> 4;
  int c0 = lane & 15;
  v8f zero = {};

  // stage V^T once (32 keys x 64 dd, rows >= 24 zeroed)
  {
    v16h v0, v1, v2, v3;
    if (lane < 24) {
      const _Float16* vr = Vp + (size_t)lane * tstride;
      v0 = *(const v16h*)(vr);
      v1 = *(const v16h*)(vr + 16);
      v2 = *(const v16h*)(vr + 32);
      v3 = *(const v16h*)(vr + 48);
    } else {
#pragma unroll
      for (int e = 0; e < 16; ++e) { v0[e] = (_Float16)0.0f; v1[e] = v0[e]; v2[e] = v0[e]; v3[e] = v0[e]; }
    }
#pragma unroll
    for (int e = 0; e < 16; ++e) {
      ldsVt[wv][(e) * 32 + lane] = v0[e];
      ldsVt[wv][(16 + e) * 32 + lane] = v1[e];
      ldsVt[wv][(32 + e) * 32 + lane] = v2[e];
      ldsVt[wv][(48 + e) * 32 + lane] = v3[e];
    }
  }
  asm volatile("s_wait_dscnt 0" ::: "memory");

  const float scale = 0.125f;
  for (int mt = 0; mt < 2; ++mt) {
    int t0 = mt * 16;
    v16h qa0 = ld_a_g(Qp, tstride, t0, 0, lane, 24);
    v16h qa1 = ld_a_g(Qp, tstride, t0, 32, lane, 24);

    v8f sc[2];
#pragma unroll
    for (int st = 0; st < 2; ++st) {
      sc[st] = zero;
      sc[st] = WMMA16(qa0, ld_b_g(Kp, tstride, st * 16, 0, lane, 24), sc[st]);
      sc[st] = WMMA16(qa1, ld_b_g(Kp, tstride, st * 16, 32, lane, 24), sc[st]);
    }
    // causal mask + scale
#pragma unroll
    for (int st = 0; st < 2; ++st)
#pragma unroll
      for (int r = 0; r < 8; ++r) {
        int tq = t0 + r + half * 8;
        int ss = st * 16 + c0;
        float v = sc[st][r] * scale;
        if (ss > tq || ss >= 24 || tq >= 24) v = NEGF;
        sc[st][r] = v;
      }
    // full softmax over 32 (masked) columns
    float mx[8];
#pragma unroll
    for (int r = 0; r < 8; ++r) mx[r] = fmaxf(sc[0][r], sc[1][r]);
#pragma unroll
    for (int d = 1; d < 16; d <<= 1)
#pragma unroll
      for (int r = 0; r < 8; ++r) mx[r] = fmaxf(mx[r], __shfl_xor(mx[r], d, 32));
    float p0[8], p1[8], rs[8];
#pragma unroll
    for (int r = 0; r < 8; ++r) {
      p0[r] = __expf(sc[0][r] - mx[r]);
      p1[r] = __expf(sc[1][r] - mx[r]);
      rs[r] = p0[r] + p1[r];
    }
#pragma unroll
    for (int d = 1; d < 16; d <<= 1)
#pragma unroll
      for (int r = 0; r < 8; ++r) rs[r] += __shfl_xor(rs[r], d, 32);
#pragma unroll
    for (int r = 0; r < 8; ++r) {
      float inv = 1.0f / rs[r];
      p0[r] *= inv; p1[r] *= inv;
      int row = r + half * 8;
      ldsP[wv][row * 32 + c0] = (_Float16)p0[r];
      ldsP[wv][row * 32 + 16 + c0] = (_Float16)p1[r];
    }
    asm volatile("s_wait_dscnt 0" ::: "memory");

    v16h pa = ld_a(&ldsP[wv][0], 32, 0, 0, lane);
    v8f acc[4];
#pragma unroll
    for (int df = 0; df < 4; ++df) {
      acc[df] = zero;
      acc[df] = WMMA16(pa, ld_b(&ldsVt[wv][0], 32, df * 16, 0, lane), acc[df]);
    }
#pragma unroll
    for (int df = 0; df < 4; ++df)
#pragma unroll
      for (int r = 0; r < 8; ++r) {
        int tq = t0 + r + half * 8;
        if (tq < 24)
          Op[(size_t)tq * tstride + df * 16 + c0] = (_Float16)acc[df][r];
      }
  }
}

// ---------------------------------------------------------------------------
// Data preparation kernels
// ---------------------------------------------------------------------------
__global__ void k_pack_xc(const float* __restrict__ X, const float* __restrict__ STE,
                          _Float16* __restrict__ Xc, long total) {
  long idx = (long)blockIdx.x * blockDim.x + threadIdx.x;
  if (idx >= total) return;
  long m = idx / 1536;
  int c = (int)(idx - m * 1536);
  float v = (c < 512) ? X[m * 512 + c] : STE[m * 1024 + (c - 512)];
  Xc[idx] = (_Float16)v;
}

__global__ void k_transpose(const float* __restrict__ W, _Float16* __restrict__ Wt,
                            int Kd, int N) {
  long idx = (long)blockIdx.x * blockDim.x + threadIdx.x;
  if (idx >= (long)Kd * N) return;
  int k = (int)(idx / N);
  int n = (int)(idx - (long)k * N);
  Wt[(size_t)n * Kd + k] = (_Float16)W[idx];
}

// ---------------------------------------------------------------------------
extern "C" void kernel_launch(void* const* d_in, const int* in_sizes, int n_in,
                              void* d_out, int out_size, void* d_ws, size_t ws_size,
                              hipStream_t stream) {
  (void)in_sizes; (void)n_in; (void)out_size; (void)ws_size;
  const float* X    = (const float*)d_in[0];
  const float* STE  = (const float*)d_in[1];
  const float* s_Wq = (const float*)d_in[2];  const float* s_bq = (const float*)d_in[3];
  const float* s_Wk = (const float*)d_in[4];  const float* s_bk = (const float*)d_in[5];
  const float* s_Wv = (const float*)d_in[6];  const float* s_bv = (const float*)d_in[7];
  const float* s_Wo = (const float*)d_in[8];  const float* s_bo = (const float*)d_in[9];
  const float* t_Wq = (const float*)d_in[10]; const float* t_bq = (const float*)d_in[11];
  const float* t_Wk = (const float*)d_in[12]; const float* t_bk = (const float*)d_in[13];
  const float* t_Wv = (const float*)d_in[14]; const float* t_bv = (const float*)d_in[15];
  const float* t_Wo = (const float*)d_in[16]; const float* t_bo = (const float*)d_in[17];
  const float* g_Wxs = (const float*)d_in[18]; const float* g_Wxt = (const float*)d_in[19];
  const float* g_bxt = (const float*)d_in[20];
  const float* g_Wh1 = (const float*)d_in[21]; const float* g_bh1 = (const float*)d_in[22];
  const float* g_Wh2 = (const float*)d_in[23]; const float* g_bh2 = (const float*)d_in[24];

  const size_t M = 24576, D = 512, D3 = 1536;
  char* ws = (char*)d_ws;
  size_t off = 0;
  auto take = [&](size_t bytes) { size_t o = off; off += (bytes + 255) & ~(size_t)255; return o; };
  size_t oXc  = take(2 * M * D3);
  size_t oWsq = take(2 * D * D3), oWsk = take(2 * D * D3), oWsv = take(2 * D * D3);
  size_t oWtq = take(2 * D * D3), oWtk = take(2 * D * D3), oWtv = take(2 * D * D3);
  size_t oWso = take(2 * D * D), oWto = take(2 * D * D), oWxs = take(2 * D * D);
  size_t oWxt = take(2 * D * D), oWh1 = take(2 * D * D), oWh2 = take(2 * D * D);
  size_t oQ = take(2 * M * D), oK = take(2 * M * D), oV = take(2 * M * D);
  size_t oAO = take(2 * M * D), oHS = take(2 * M * D), oHT = take(2 * M * D);
  size_t oG1 = oQ;   // f32 M*D fits exactly in Q+K region (both dead by then)
  size_t oHf = oV;   // dead after temporal attention
  size_t oH1 = oAO;  // dead after HT projection

  _Float16* Xc  = (_Float16*)(ws + oXc);
  _Float16* Wsq = (_Float16*)(ws + oWsq); _Float16* Wsk = (_Float16*)(ws + oWsk);
  _Float16* Wsv = (_Float16*)(ws + oWsv); _Float16* Wtq = (_Float16*)(ws + oWtq);
  _Float16* Wtk = (_Float16*)(ws + oWtk); _Float16* Wtv = (_Float16*)(ws + oWtv);
  _Float16* Wso = (_Float16*)(ws + oWso); _Float16* Wto = (_Float16*)(ws + oWto);
  _Float16* Wxs = (_Float16*)(ws + oWxs); _Float16* Wxt = (_Float16*)(ws + oWxt);
  _Float16* Wh1 = (_Float16*)(ws + oWh1); _Float16* Wh2 = (_Float16*)(ws + oWh2);
  _Float16* Qb = (_Float16*)(ws + oQ); _Float16* Kb = (_Float16*)(ws + oK);
  _Float16* Vb = (_Float16*)(ws + oV);
  _Float16* AO = (_Float16*)(ws + oAO);
  _Float16* HS = (_Float16*)(ws + oHS); _Float16* HT = (_Float16*)(ws + oHT);
  float* G1 = (float*)(ws + oG1);
  _Float16* Hf = (_Float16*)(ws + oHf);
  _Float16* H1 = (_Float16*)(ws + oH1);

  dim3 blk(256);
  long packN = (long)M * D3;
  k_pack_xc<<<(unsigned)((packN + 255) / 256), blk, 0, stream>>>(X, STE, Xc, packN);

  unsigned gBig = (unsigned)((D3 * D + 255) / 256);
  unsigned gSml = (unsigned)((D * D + 255) / 256);
  k_transpose<<<gBig, blk, 0, stream>>>(s_Wq, Wsq, (int)D3, (int)D);
  k_transpose<<<gBig, blk, 0, stream>>>(s_Wk, Wsk, (int)D3, (int)D);
  k_transpose<<<gBig, blk, 0, stream>>>(s_Wv, Wsv, (int)D3, (int)D);
  k_transpose<<<gBig, blk, 0, stream>>>(t_Wq, Wtq, (int)D3, (int)D);
  k_transpose<<<gBig, blk, 0, stream>>>(t_Wk, Wtk, (int)D3, (int)D);
  k_transpose<<<gBig, blk, 0, stream>>>(t_Wv, Wtv, (int)D3, (int)D);
  k_transpose<<<gSml, blk, 0, stream>>>(s_Wo, Wso, (int)D, (int)D);
  k_transpose<<<gSml, blk, 0, stream>>>(t_Wo, Wto, (int)D, (int)D);
  k_transpose<<<gSml, blk, 0, stream>>>(g_Wxs, Wxs, (int)D, (int)D);
  k_transpose<<<gSml, blk, 0, stream>>>(g_Wxt, Wxt, (int)D, (int)D);
  k_transpose<<<gSml, blk, 0, stream>>>(g_Wh1, Wh1, (int)D, (int)D);
  k_transpose<<<gSml, blk, 0, stream>>>(g_Wh2, Wh2, (int)D, (int)D);

  unsigned gGemm = (unsigned)((M / 64) * (D / 256));  // 768

  // spatial QKV -> attention -> HS
  k_gemm_relu_f16<<<gGemm, blk, 0, stream>>>(Xc, (int)D3, Wsq, (int)D3, s_bq, Qb, (int)D3);
  k_gemm_relu_f16<<<gGemm, blk, 0, stream>>>(Xc, (int)D3, Wsk, (int)D3, s_bk, Kb, (int)D3);
  k_gemm_relu_f16<<<gGemm, blk, 0, stream>>>(Xc, (int)D3, Wsv, (int)D3, s_bv, Vb, (int)D3);
  k_sp_attn<<<1536, blk, 0, stream>>>(Qb, Kb, Vb, AO);
  k_gemm_relu_f16<<<gGemm, blk, 0, stream>>>(AO, (int)D, Wso, (int)D, s_bo, HS, (int)D);

  // temporal QKV (reuse buffers) -> attention -> HT
  k_gemm_relu_f16<<<gGemm, blk, 0, stream>>>(Xc, (int)D3, Wtq, (int)D3, t_bq, Qb, (int)D3);
  k_gemm_relu_f16<<<gGemm, blk, 0, stream>>>(Xc, (int)D3, Wtk, (int)D3, t_bk, Kb, (int)D3);
  k_gemm_relu_f16<<<gGemm, blk, 0, stream>>>(Xc, (int)D3, Wtv, (int)D3, t_bv, Vb, (int)D3);
  k_tm_attn<<<1024, blk, 0, stream>>>(Qb, Kb, Vb, AO);
  k_gemm_relu_f16<<<gGemm, blk, 0, stream>>>(AO, (int)D, Wto, (int)D, t_bo, HT, (int)D);

  // gated fusion + MLP + residual
  k_gemm_f32<<<gGemm, blk, 0, stream>>>(HS, (int)D, Wxs, (int)D, G1, (int)D);
  k_fuse<<<gGemm, blk, 0, stream>>>(HT, (int)D, Wxt, (int)D, g_bxt, G1, HS, HT, Hf, (int)D);
  k_gemm_relu_f16<<<gGemm, blk, 0, stream>>>(Hf, (int)D, Wh1, (int)D, g_bh1, H1, (int)D);
  k_final<<<gGemm, blk, 0, stream>>>(H1, (int)D, Wh2, (int)D, g_bh2, X, (float*)d_out, (int)D);
}
